// CAMoEBlock_13692355739771
// MI455X (gfx1250) — matrix operand order repaired and checked by reading, not compiled, not run on hardware
//
#include <hip/hip_runtime.h>
#include <hip/hip_bf16.h>
#include <math.h>
#include <stdint.h>

typedef _Float16 half_t;
typedef __attribute__((ext_vector_type(16))) _Float16 v16h;
typedef __attribute__((ext_vector_type(8)))  _Float16 v8h;
typedef __attribute__((ext_vector_type(8)))  float    v8f;

#define TB 2
#define TS 1024
#define TD 1024
#define TH 16
#define THD 64
#define TE 8
#define TF 2048
#define TT (TB * TS)   // 2048 tokens

__device__ __forceinline__ v8f wmma16(v16h a, v16h b, v8f c) {
  return __builtin_amdgcn_wmma_f32_16x16x32_f16(false, a, false, b, (short)0, c, false, false);
}

__device__ __forceinline__ v8f zero8() {
  v8f c;
#pragma unroll
  for (int i = 0; i < 8; ++i) c[i] = 0.f;
  return c;
}

// Async global->LDS 16-byte copy (per lane). dst: LDS pointer, src: global.
__device__ __forceinline__ void async_copy16(const half_t* lds_dst, const half_t* gsrc) {
  unsigned l = (unsigned)(uintptr_t)lds_dst;                 // LDS address: low 32 bits of the pointer
  unsigned long long g = (unsigned long long)(uintptr_t)gsrc;
  asm volatile("global_load_async_to_lds_b128 %0, %1, off" :: "v"(l), "v"(g) : "memory");
}

__device__ __forceinline__ void async_wait0() {
  asm volatile("s_wait_asynccnt 0" ::: "memory");
}

// ---------------------------------------------------------------------------
// Weight conversion: f32 -> f16 (plain), 4 elements per thread.
// ---------------------------------------------------------------------------
__global__ void conv_kernel(const float* __restrict__ in, half_t* __restrict__ out) {
  size_t i = ((size_t)blockIdx.x * 256 + threadIdx.x) * 4;
  float4 v = *(const float4*)(in + i);
  out[i + 0] = (half_t)v.x;
  out[i + 1] = (half_t)v.y;
  out[i + 2] = (half_t)v.z;
  out[i + 3] = (half_t)v.w;
}

// ---------------------------------------------------------------------------
// Weight conversion with transpose: in [R][C] f32 -> out [C][R] f16.
// blockIdx.z = expert slab. Tiled 32x32 through LDS, block (32,8).
// ---------------------------------------------------------------------------
__global__ void convT_kernel(const float* __restrict__ in, half_t* __restrict__ out,
                             int R, int C) {
  __shared__ float tile[32][33];
  const float* inp = in + (size_t)blockIdx.z * R * C;
  half_t* outp = out + (size_t)blockIdx.z * R * C;
  int r0 = blockIdx.y * 32, c0 = blockIdx.x * 32;
  int tx = threadIdx.x, ty = threadIdx.y;
#pragma unroll
  for (int i = 0; i < 32; i += 8)
    tile[ty + i][tx] = inp[(size_t)(r0 + ty + i) * C + c0 + tx];
  __syncthreads();
#pragma unroll
  for (int i = 0; i < 32; i += 8)
    outp[(size_t)(c0 + ty + i) * R + r0 + tx] = (half_t)tile[tx][ty + i];
}

// ---------------------------------------------------------------------------
// LayerNorm (D=1024), block = 256 threads, one block per token. Output f16.
// ---------------------------------------------------------------------------
__global__ void ln_kernel(const float* __restrict__ x, const float* __restrict__ g,
                          const float* __restrict__ b, half_t* __restrict__ out) {
  const int D = TD;
  int t = blockIdx.x;
  const float* row = x + (size_t)t * D;
  __shared__ float red[16];
  __shared__ float mv[2];
  float s = 0.f, s2 = 0.f;
  for (int i = threadIdx.x; i < D; i += 256) { float v = row[i]; s += v; s2 += v * v; }
#pragma unroll
  for (int off = 16; off > 0; off >>= 1) {
    s  += __shfl_xor(s,  off, 32);
    s2 += __shfl_xor(s2, off, 32);
  }
  int w = threadIdx.x >> 5;
  if ((threadIdx.x & 31) == 0) { red[w] = s; red[8 + w] = s2; }
  __syncthreads();
  if (threadIdx.x == 0) {
    float a = 0.f, c = 0.f;
    for (int i = 0; i < 8; ++i) { a += red[i]; c += red[8 + i]; }
    float mu = a / D;
    mv[0] = mu;
    mv[1] = rsqrtf(c / D - mu * mu + 1e-5f);
  }
  __syncthreads();
  float mu = mv[0], rstd = mv[1];
  half_t* orow = out + (size_t)t * D;
  for (int i = threadIdx.x; i < D; i += 256)
    orow[i] = (half_t)((row[i] - mu) * rstd * g[i] + b[i]);
}

// ---------------------------------------------------------------------------
// WMMA GEMM: C[M,N] = A_f16[M,K] @ W16^T (W16 f16 [N,K] row-major) + bias.
// Block = 8 waves sharing one 16-row A tile async-staged in LDS; each wave a
// 16x64 N-slice. K == 1024 in all uses; grids sized exactly.
// mode 0: store f16. mode 1: f32 + residual.
// ---------------------------------------------------------------------------
__global__ void gemm_awT_kernel(const half_t* __restrict__ A, const half_t* __restrict__ W,
                                const float* __restrict__ bias,
                                half_t* __restrict__ out16, float* __restrict__ out32,
                                const float* __restrict__ resid,
                                int M, int N, int K, int mode) {
  __shared__ __align__(32) half_t aTile[16 * 1024];
  int wv   = threadIdx.x >> 5;
  int lane = threadIdx.x & 31;
  int ntiles = N >> 6;
  int gw0 = blockIdx.x * 8;
  int mt  = gw0 / ntiles;
  int nt  = (gw0 % ntiles) + wv;

  // async stage of A tile (16 x K halves)
  for (int i = threadIdx.x; i < 16 * (K / 8); i += 256) {
    int row = i / (K / 8);
    int c8  = (i - row * (K / 8)) * 8;
    async_copy16(&aTile[row * K + c8], &A[(size_t)(mt * 16 + row) * K + c8]);
  }
  async_wait0();
  __syncthreads();

  v8f c[4];
#pragma unroll
  for (int s = 0; s < 4; ++s) c[s] = zero8();

  int colb = lane & 15;
  int kb   = (lane >> 4) * 16;
  int kbB  = (lane >> 4) * 16;

  for (int k0 = 0; k0 < K; k0 += 32) {
    v16h a;
    {
      const half_t* p = aTile + (lane & 15) * K + k0 + kb;
      v8h lo = *(const v8h*)p;
      v8h hi = *(const v8h*)(p + 8);
#pragma unroll
      for (int i = 0; i < 8; ++i) { a[i] = lo[i]; a[8 + i] = hi[i]; }
    }
#pragma unroll
    for (int s = 0; s < 4; ++s) {
      int col = nt * 64 + s * 16 + colb;
      v16h bf = *(const v16h*)(W + (size_t)col * K + k0 + kbB);
      c[s] = wmma16(a, bf, c[s]);
    }
  }

  int rbase = (lane >> 4) * 8;
#pragma unroll
  for (int s = 0; s < 4; ++s) {
    int col = nt * 64 + s * 16 + colb;
    float bb = bias[col];
#pragma unroll
    for (int r = 0; r < 8; ++r) {
      size_t idx = (size_t)(mt * 16 + rbase + r) * N + col;
      float v = c[s][r] + bb;
      if (mode == 0) out16[idx] = (half_t)v;
      else           out32[idx] = v + resid[idx];
    }
  }
}

// ---------------------------------------------------------------------------
// Flash attention: one block (8 waves) per (b, h, 128-query chunk).
// K/V tiles (32x64 f16) double-buffered in LDS via async DMA; the next tile's
// copy overlaps the current tile's WMMA work.
// ---------------------------------------------------------------------------
__global__ void attn_kernel(const half_t* __restrict__ qkv, half_t* __restrict__ ctx) {
  __shared__ __align__(32) half_t lK[2][32 * 64];
  __shared__ __align__(32) half_t lV[2][32 * 64];
  __shared__ __align__(32) half_t pshare[8][16 * 32];
  int wv   = threadIdx.x >> 5;
  int lane = threadIdx.x & 31;
  int blk = blockIdx.x;                 // 0..255
  int qc = blk & 7;
  int h  = (blk >> 3) & 15;
  int b  = blk >> 7;
  int qt = qc * 8 + wv;                 // this wave's 16-query tile

  size_t base = (size_t)b * TS * (3 * TD);
  const half_t* Qg = qkv + base + h * THD;
  const half_t* Kg = qkv + base + TD + h * THD;
  const half_t* Vg = qkv + base + 2 * TD + h * THD;

  // per-thread staging slot: 8 halves (16 bytes)
  int sidx = threadIdx.x * 8;
  int srow = sidx >> 6, scol = sidx & 63;

  // Q fragments (held for whole loop)
  int qrow = qt * 16 + (lane & 15);
  int kb   = (lane >> 4) * 16;
  const half_t* qp = Qg + (size_t)qrow * (3 * TD);
  v16h aq0, aq1;
#pragma unroll
  for (int i = 0; i < 8; ++i) {
    aq0[i]     = qp[kb + i];
    aq0[8 + i] = qp[kb + 8 + i];
    aq1[i]     = qp[32 + kb + i];
    aq1[8 + i] = qp[32 + kb + 8 + i];
  }

  float m[8], l[8];
  v8f o[4];
#pragma unroll
  for (int r = 0; r < 8; ++r) { m[r] = -1e30f; l[r] = 0.f; }
#pragma unroll
  for (int s = 0; s < 4; ++s) o[s] = zero8();

  half_t* pb = pshare[wv];
  const float scale = 0.125f;           // 1/sqrt(64)
  int colb = lane & 15;
  int kbB  = (lane >> 4) * 16;

  // prologue: stage first K/V tile into buffer 0
  async_copy16(&lK[0][sidx], &Kg[(size_t)srow * (3 * TD) + scol]);
  async_copy16(&lV[0][sidx], &Vg[(size_t)srow * (3 * TD) + scol]);
  async_wait0();
  __syncthreads();

  int p = 0;
  for (int kt = 0; kt < TS; kt += 32) {
    // issue async copy of next tile into the other buffer (overlaps compute)
    if (kt + 32 < TS) {
      async_copy16(&lK[p ^ 1][sidx], &Kg[(size_t)(kt + 32 + srow) * (3 * TD) + scol]);
      async_copy16(&lV[p ^ 1][sidx], &Vg[(size_t)(kt + 32 + srow) * (3 * TD) + scol]);
    }

    // ---- scores: 16x32 tile (two C fragments) ----
    v8f s0 = zero8(), s1 = zero8();
    {
      const half_t* kp = lK[p] + colb * 64;
      s0 = wmma16(aq0, *(const v16h*)(kp + kbB), s0);
      s0 = wmma16(aq1, *(const v16h*)(kp + 32 + kbB), s0);
      const half_t* kp2 = lK[p] + (16 + colb) * 64;
      s1 = wmma16(aq0, *(const v16h*)(kp2 + kbB), s1);
      s1 = wmma16(aq1, *(const v16h*)(kp2 + 32 + kbB), s1);
    }
    // ---- online softmax update ----
#pragma unroll
    for (int r = 0; r < 8; ++r) {
      float v = fmaxf(s0[r], s1[r]) * scale;
#pragma unroll
      for (int off = 8; off > 0; off >>= 1) v = fmaxf(v, __shfl_xor(v, off, 16));
      float mnew = fmaxf(m[r], v);
      float p0 = __expf(s0[r] * scale - mnew);
      float p1 = __expf(s1[r] * scale - mnew);
      s0[r] = p0; s1[r] = p1;
      float rs = p0 + p1;
#pragma unroll
      for (int off = 8; off > 0; off >>= 1) rs += __shfl_xor(rs, off, 16);
      float corr = __expf(m[r] - mnew);
      l[r] = l[r] * corr + rs;
      m[r] = mnew;
#pragma unroll
      for (int s = 0; s < 4; ++s) o[s][r] *= corr;
    }
    // ---- transpose P through LDS into A-fragment layout ----
    int prow = (lane >> 4) * 8;
#pragma unroll
    for (int r = 0; r < 8; ++r) {
      pb[(prow + r) * 32 + colb]      = (half_t)s0[r];
      pb[(prow + r) * 32 + 16 + colb] = (half_t)s1[r];
    }
    v16h ap;
    {
      const half_t* pp = pb + (lane & 15) * 32 + kb;
      v8h lo = *(const v8h*)pp;
      v8h hi = *(const v8h*)(pp + 8);
#pragma unroll
      for (int i = 0; i < 8; ++i) { ap[i] = lo[i]; ap[8 + i] = hi[i]; }
    }
    // ---- O += P @ V (32 keys x 64 dims -> 4 WMMAs) ----
#pragma unroll
    for (int s = 0; s < 4; ++s) {
      v16h bv;
#pragma unroll
      for (int i = 0; i < 16; ++i)
        bv[i] = lV[p][(kbB + i) * 64 + s * 16 + colb];
      o[s] = wmma16(ap, bv, o[s]);
    }

    async_wait0();       // next tile's DMA has landed
    __syncthreads();     // everyone done reading buffer p
    p ^= 1;
  }

  // ---- epilogue: ctx = O / l ----
  int rbase = (lane >> 4) * 8;
#pragma unroll
  for (int s = 0; s < 4; ++s) {
#pragma unroll
    for (int r = 0; r < 8; ++r) {
      int sr = qt * 16 + rbase + r;
      float v = o[s][r] / l[r];
      ctx[(size_t)(b * TS + sr) * TD + h * THD + s * 16 + colb] = (half_t)v;
    }
  }
}

// ---------------------------------------------------------------------------
// Router: one wave per token. Writes top-2 indices + normalized gates.
// ---------------------------------------------------------------------------
__global__ void router_kernel(const half_t* __restrict__ xn, const float* __restrict__ rw,
                              const float* __restrict__ rb,
                              int* __restrict__ topi, float* __restrict__ topg) {
  int lane = threadIdx.x & 31;
  int t = (blockIdx.x * blockDim.x + threadIdx.x) >> 5;
  const half_t* x = xn + (size_t)t * TD;
  float logit[TE];
#pragma unroll
  for (int e = 0; e < TE; ++e) logit[e] = 0.f;
  for (int d = lane; d < TD; d += 32) {
    float xv = (float)x[d];
    const float* w = rw + (size_t)d * TE;
#pragma unroll
    for (int e = 0; e < TE; ++e) logit[e] += xv * w[e];
  }
#pragma unroll
  for (int e = 0; e < TE; ++e) {
#pragma unroll
    for (int off = 16; off > 0; off >>= 1) logit[e] += __shfl_xor(logit[e], off, 32);
    logit[e] += rb[e];
  }
  float mx = logit[0];
#pragma unroll
  for (int e = 1; e < TE; ++e) mx = fmaxf(mx, logit[e]);
  float p[TE]; float se = 0.f;
#pragma unroll
  for (int e = 0; e < TE; ++e) { p[e] = __expf(logit[e] - mx); se += p[e]; }
#pragma unroll
  for (int e = 0; e < TE; ++e) p[e] /= se;
  int i0 = 0;
#pragma unroll
  for (int e = 1; e < TE; ++e) if (p[e] > p[i0]) i0 = e;
  int i1 = (i0 == 0) ? 1 : 0;
#pragma unroll
  for (int e = 0; e < TE; ++e) if (e != i0 && p[e] > p[i1]) i1 = e;
  float g0 = p[i0], g1 = p[i1], gs = g0 + g1;
  if (lane == 0) {
    topi[t * 2] = i0; topi[t * 2 + 1] = i1;
    topg[t * 2] = g0 / gs; topg[t * 2 + 1] = g1 / gs;
  }
}

// ---------------------------------------------------------------------------
// Deterministic compaction: lane e scans all tokens, builds per-expert lists.
// ---------------------------------------------------------------------------
__global__ void compact_kernel(const int* __restrict__ topi, const float* __restrict__ topg,
                               int* __restrict__ counts, int* __restrict__ tlist,
                               float* __restrict__ glist, int* __restrict__ rlist) {
  int e = threadIdx.x;
  if (e >= TE) return;
  int c = 0;
  for (int t = 0; t < TT; ++t) {
    int i0 = topi[t * 2], i1 = topi[t * 2 + 1];
    if (i0 == e) { tlist[e * TT + c] = t; glist[e * TT + c] = topg[t * 2];     rlist[e * TT + c] = 0; ++c; }
    else if (i1 == e) { tlist[e * TT + c] = t; glist[e * TT + c] = topg[t * 2 + 1]; rlist[e * TT + c] = 1; ++c; }
  }
  counts[e] = c;
}

// ---------------------------------------------------------------------------
// Fused MoE: one workgroup (8 waves) per (expert, 16-token tile).
// w1t: [E][F][D] f16 (transposed), w2t: [E][D][F] f16 (transposed).
// GEMM1 + GeLU -> LDS mid (F split into 2 chunks of 1024), GEMM2 -> rank buf.
// ---------------------------------------------------------------------------
__global__ void moe_kernel(const half_t* __restrict__ xn2,
                           const half_t* __restrict__ w1t, const float* __restrict__ b1,
                           const half_t* __restrict__ w2t, const float* __restrict__ b2,
                           const int* __restrict__ counts, const int* __restrict__ tlist,
                           const float* __restrict__ glist, const int* __restrict__ rlist,
                           float* __restrict__ moe0, float* __restrict__ moe1) {
  __shared__ __align__(32) half_t mid[16 * 1024];
  __shared__ int   toks[16];
  __shared__ float gts[16];
  __shared__ int   rks[16];
  __shared__ int   s_cnt;

  int e = blockIdx.y;
  int tile = blockIdx.x;
  if (threadIdx.x == 0) s_cnt = counts[e];
  __syncthreads();
  int cnt = s_cnt;
  if (tile * 16 >= cnt) return;

  if (threadIdx.x < 16) {
    int slot = tile * 16 + threadIdx.x;
    int idx = (slot < cnt) ? slot : tile * 16;
    toks[threadIdx.x] = tlist[e * TT + idx];
    gts[threadIdx.x]  = (slot < cnt) ? glist[e * TT + slot] : 0.f;
    rks[threadIdx.x]  = (slot < cnt) ? rlist[e * TT + slot] : 0;
  }
  __syncthreads();

  int wv   = threadIdx.x >> 5;
  int lane = threadIdx.x & 31;
  int colb = lane & 15;
  int kb   = (lane >> 4) * 16;
  int kbB  = (lane >> 4) * 16;
  int rbase = (lane >> 4) * 8;

  const half_t* w1te = w1t + (size_t)e * TD * TF;
  const half_t* w2te = w2t + (size_t)e * TF * TD;
  const float* b1e = b1 + (size_t)e * TF;
  const float* b2e = b2 + (size_t)e * TD;

  v8f acc2[8];
#pragma unroll
  for (int q = 0; q < 8; ++q) acc2[q] = zero8();

  for (int fc = 0; fc < 2; ++fc) {
    // ---- GEMM1: mid_chunk[16][1024] = gelu(x @ w1[:, fc*1024 : +1024] + b1) ----
    v8f acc1[8];
#pragma unroll
    for (int q = 0; q < 8; ++q) acc1[q] = zero8();

    for (int k0 = 0; k0 < TD; k0 += 32) {
      v16h a;
      {
        const half_t* xp = xn2 + (size_t)toks[lane & 15] * TD + k0 + kb;
#pragma unroll
        for (int i = 0; i < 8; ++i) { a[i] = xp[i]; a[8 + i] = xp[8 + i]; }
      }
#pragma unroll
      for (int q = 0; q < 8; ++q) {
        int fcol = fc * 1024 + (wv * 8 + q) * 16 + colb;
        v16h bf = *(const v16h*)(w1te + (size_t)fcol * TD + k0 + kbB);
        acc1[q] = wmma16(a, bf, acc1[q]);
      }
    }

    __syncthreads();   // previous chunk's GEMM2 reads are done
#pragma unroll
    for (int q = 0; q < 8; ++q) {
      int lcol = (wv * 8 + q) * 16 + colb;
      float bb = b1e[fc * 1024 + lcol];
#pragma unroll
      for (int r = 0; r < 8; ++r) {
        float v = acc1[q][r] + bb;
        v = 0.5f * v * (1.f + erff(v * 0.70710678f));
        mid[(rbase + r) * 1024 + lcol] = (half_t)v;
      }
    }
    __syncthreads();

    // ---- GEMM2 partial: accumulate over this 1024 K-chunk ----
    for (int k0 = 0; k0 < 1024; k0 += 32) {
      v16h a;
      {
        const half_t* mp = mid + (lane & 15) * 1024 + k0 + kb;
        v8h lo = *(const v8h*)mp;
        v8h hi = *(const v8h*)(mp + 8);
#pragma unroll
        for (int i = 0; i < 8; ++i) { a[i] = lo[i]; a[8 + i] = hi[i]; }
      }
#pragma unroll
      for (int q = 0; q < 8; ++q) {
        int dcol = (wv * 8 + q) * 16 + colb;
        v16h bf = *(const v16h*)(w2te + (size_t)dcol * TF + fc * 1024 + k0 + kbB);
        acc2[q] = wmma16(a, bf, acc2[q]);
      }
    }
  }

  // ---- epilogue: write gate-scaled expert output into rank buffer ----
#pragma unroll
  for (int q = 0; q < 8; ++q) {
    int dcol = (wv * 8 + q) * 16 + colb;
    float bb = b2e[dcol];
#pragma unroll
    for (int r = 0; r < 8; ++r) {
      int slot = rbase + r;
      if (tile * 16 + slot < cnt) {
        float v = (acc2[q][r] + bb) * gts[slot];
        float* dst = (rks[slot] == 0) ? moe0 : moe1;
        dst[(size_t)toks[slot] * TD + dcol] = v;
      }
    }
  }
}

// ---------------------------------------------------------------------------
// Combine: out = h + moe_rank0 + moe_rank1
// ---------------------------------------------------------------------------
__global__ void combine_kernel(const float* __restrict__ h, const float* __restrict__ m0,
                               const float* __restrict__ m1, float* __restrict__ out) {
  size_t i = (size_t)blockIdx.x * blockDim.x + threadIdx.x;
  out[i] = h[i] + m0[i] + m1[i];
}

// ---------------------------------------------------------------------------
extern "C" void kernel_launch(void* const* d_in, const int* in_sizes, int n_in,
                              void* d_out, int out_size, void* d_ws, size_t ws_size,
                              hipStream_t stream) {
  (void)in_sizes; (void)n_in; (void)out_size; (void)ws_size;
  const float* hs   = (const float*)d_in[0];
  const float* ln1g = (const float*)d_in[1];
  const float* ln1b = (const float*)d_in[2];
  const float* wqkv = (const float*)d_in[3];
  const float* bqkv = (const float*)d_in[4];
  const float* wo   = (const float*)d_in[5];
  const float* bo   = (const float*)d_in[6];
  const float* ln2g = (const float*)d_in[7];
  const float* ln2b = (const float*)d_in[8];
  const float* rw   = (const float*)d_in[9];
  const float* rb   = (const float*)d_in[10];
  const float* w1   = (const float*)d_in[11];
  const float* b1   = (const float*)d_in[12];
  const float* w2   = (const float*)d_in[13];
  const float* b2   = (const float*)d_in[14];
  float* out = (float*)d_out;

  // workspace layout (all section sizes are multiples of 32 bytes)
  char* ws = (char*)d_ws;
  half_t* xn1    = (half_t*)ws;                       // T*D f16
  half_t* qkv16  = xn1 + (size_t)TT * TD;             // T*3D f16
  half_t* ctx16  = qkv16 + (size_t)TT * 3 * TD;       // T*D f16
  half_t* xn2    = ctx16 + (size_t)TT * TD;           // T*D f16
  float*  h32    = (float*)(xn2 + (size_t)TT * TD);   // T*D f32
  float*  moe0   = h32 + (size_t)TT * TD;             // T*D f32
  float*  moe1   = moe0 + (size_t)TT * TD;            // T*D f32
  int*    topi   = (int*)(moe1 + (size_t)TT * TD);    // T*2
  float*  topg   = (float*)(topi + TT * 2);           // T*2
  int*    cnts   = (int*)(topg + TT * 2);             // E
  int*    tlist  = cnts + TE;                         // E*T
  float*  glist  = (float*)(tlist + TE * TT);         // E*T
  int*    rlist  = (int*)(glist + TE * TT);           // E*T
  half_t* wqkv16 = (half_t*)(rlist + TE * TT);        // 3D*D f16
  half_t* wo16   = wqkv16 + (size_t)3 * TD * TD;      // D*D f16
  half_t* w1t    = wo16 + (size_t)TD * TD;            // E*F*D f16 (transposed)
  half_t* w2t    = w1t + (size_t)TE * TD * TF;        // E*D*F f16 (transposed)

  // 0. weight conversion (f32 -> f16; experts transposed for contiguous K)
  conv_kernel<<<(3 * TD * TD) / 1024, 256, 0, stream>>>(wqkv, wqkv16);
  conv_kernel<<<(TD * TD) / 1024, 256, 0, stream>>>(wo, wo16);
  convT_kernel<<<dim3(TF / 32, TD / 32, TE), dim3(32, 8), 0, stream>>>(w1, w1t, TD, TF);
  convT_kernel<<<dim3(TD / 32, TF / 32, TE), dim3(32, 8), 0, stream>>>(w2, w2t, TF, TD);
  // 1. LN1
  ln_kernel<<<TT, 256, 0, stream>>>(hs, ln1g, ln1b, xn1);
  // 2. QKV GEMM: [2048,1024] x [1024,3072]
  gemm_awT_kernel<<<(TT / 16) * (3 * TD / 64) / 8, 256, 0, stream>>>(
      xn1, wqkv16, bqkv, qkv16, nullptr, nullptr, TT, 3 * TD, TD, 0);
  // 3. attention: one block per (b, h, 128 queries)
  attn_kernel<<<TB * TH * (TS / 128), 256, 0, stream>>>(qkv16, ctx16);
  // 4. o-proj + residual
  gemm_awT_kernel<<<(TT / 16) * (TD / 64) / 8, 256, 0, stream>>>(
      ctx16, wo16, bo, nullptr, h32, hs, TT, TD, TD, 1);
  // 5. LN2
  ln_kernel<<<TT, 256, 0, stream>>>(h32, ln2g, ln2b, xn2);
  // 6. router: one wave per token
  router_kernel<<<TT / 8, 256, 0, stream>>>(xn2, rw, rb, topi, topg);
  // 7. deterministic compaction
  compact_kernel<<<1, 32, 0, stream>>>(topi, topg, cnts, tlist, glist, rlist);
  // 8. fused MoE
  moe_kernel<<<dim3(TT / 16, TE), 256, 0, stream>>>(
      xn2, w1t, b1, w2t, b2, cnts, tlist, glist, rlist, moe0, moe1);
  // 9. combine
  combine_kernel<<<(TT * TD) / 256, 256, 0, stream>>>(h32, moe0, moe1, out);
}
